// VQVAEASR_84207128805804
// MI455X (gfx1250) — matrix operand order repaired and checked by reading, not compiled, not run on hardware
//
#include <hip/hip_runtime.h>
#include <hip/hip_bf16.h>
#include <math.h>

// ---------------------------------------------------------------------------
// VQ-VAE ASR transformer forward for MI455X (gfx1250, wave32, WMMA).
// All dense GEMMs run on v_wmma_f32_16x16x32_bf16 (f32 accumulate).
// ---------------------------------------------------------------------------

typedef __bf16 bf16;
typedef __attribute__((ext_vector_type(16))) __bf16 v16bf;
typedef __attribute__((ext_vector_type(4)))  __bf16 v4bf;
typedef __attribute__((ext_vector_type(8)))  float  v8f;

#if defined(__has_builtin)
#  if __has_builtin(__builtin_amdgcn_sched_barrier)
#    define SCHED_FENCE() __builtin_amdgcn_sched_barrier(0)
#  endif
#endif
#ifndef SCHED_FENCE
#  define SCHED_FENCE()
#endif

#define CB   16     // batch
#define CS   300    // encoder seq
#define CD   768    // model dim
#define CK   512    // codebook size
#define CV   5000   // vocab
#define CH   8      // heads
#define CHD  96     // head dim
#define CT   49     // decoder seq
#define CL   50     // label length
#define CF   1536   // ffn dim

#define BM 64
#define BN 64
#define BK 32

// ---------------------------------------------------------------------------
// Generic tiled WMMA GEMM:  C = alpha * (A @ op(B)) + bias, optional ReLU.
//   A: M x Kd, row-major, leading dim lda (f32)
//   B: if transB: N x Kd row-major (computes A@B^T); else Kd x N row-major.
//   Batched via blockIdx.z with (outer,inner) stride pairs.
// Fast staging path: unguarded float4 loads + packed bf16x4 LDS stores for
// interior tiles. Edge fallback: two-phase — batch-issue 16 clamped loads
// into registers (sched_barrier keeps them together => one wait), then
// mask-multiply + convert + LDS store.
// ---------------------------------------------------------------------------
__global__ __launch_bounds__(128)
void gemm_wmma(const float* __restrict__ A, long long lda, long long sAo, long long sAi,
               const float* __restrict__ Bm, long long ldb, long long sBo, long long sBi,
               float* __restrict__ C, long long ldc, long long sCo, long long sCi,
               int M, int N, int Kd, int inner,
               const float* __restrict__ bias, float alpha, int relu, int transB)
{
    __shared__ bf16 As[BM][BK];       // row stride 64 B (b128-friendly)
    __shared__ bf16 Bs[BK][BN + 8];   // row stride 144 B; frag bases 16B-aligned

    const int z = blockIdx.z;
    const float* Ab = A  + (long long)(z / inner) * sAo + (long long)(z % inner) * sAi;
    const float* Bb = Bm + (long long)(z / inner) * sBo + (long long)(z % inner) * sBi;
    float*       Cb = C  + (long long)(z / inner) * sCo + (long long)(z % inner) * sCi;

    const int ldai = (int)lda, ldbi = (int)ldb;
    const int m0  = blockIdx.y * BM;
    const int n0  = blockIdx.x * BN;
    const int tid = threadIdx.x;
    const int lane = tid & 31;
    const int wave = tid >> 5;

    const bool fullM = (m0 + BM <= M);
    const bool fullN = (n0 + BN <= N);
    const bool avec = ((((unsigned long long)Ab) & 15ull) == 0) && ((ldai & 3) == 0);
    const bool bvec = ((((unsigned long long)Bb) & 15ull) == 0) && ((ldbi & 3) == 0);

    v8f acc[4] = {};

    for (int k0 = 0; k0 < Kd; k0 += BK) {
        const bool fullK = (k0 + BK <= Kd);

        // ---------------- stage A tile (64 x 32) ----------------
        if (fullM && fullK && avec) {
            #pragma unroll
            for (int r = 0; r < 4; ++r) {
                int t  = tid + r * 128;          // 0..511
                int m  = t >> 3;                 // 8 float4 per row
                int k4 = (t & 7) << 2;
                const float4 v = *(const float4*)(Ab + (size_t)((m0 + m) * ldai + k0 + k4));
                v4bf p = { (bf16)v.x, (bf16)v.y, (bf16)v.z, (bf16)v.w };
                *(v4bf*)&As[m][k4] = p;
            }
        } else {
            float vals[16];
            #pragma unroll
            for (int r = 0; r < 16; ++r) {
                int t = tid + r * 128;           // 0..2047
                int m = t >> 5, k = t & 31;
                int cm = (m0 + m) < M  ? (m0 + m) : M - 1;
                int ck = (k0 + k) < Kd ? (k0 + k) : Kd - 1;
                vals[r] = Ab[(size_t)(cm * ldai + ck)];      // always in range
            }
            SCHED_FENCE();
            #pragma unroll
            for (int r = 0; r < 16; ++r) {
                int t = tid + r * 128;
                int m = t >> 5, k = t & 31;
                float msk = ((m0 + m) < M && (k0 + k) < Kd) ? 1.f : 0.f;
                As[m][k] = (bf16)(vals[r] * msk);
            }
        }

        // ---------------- stage B tile as [k][n] ----------------
        if (transB) {
            if (fullN && fullK && bvec) {
                #pragma unroll
                for (int r = 0; r < 4; ++r) {
                    int t  = tid + r * 128;
                    int n  = t >> 3;
                    int k4 = (t & 7) << 2;
                    const float4 v = *(const float4*)(Bb + (size_t)((n0 + n) * ldbi + k0 + k4));
                    Bs[k4 + 0][n] = (bf16)v.x;
                    Bs[k4 + 1][n] = (bf16)v.y;
                    Bs[k4 + 2][n] = (bf16)v.z;
                    Bs[k4 + 3][n] = (bf16)v.w;
                }
            } else {
                float vals[16];
                #pragma unroll
                for (int r = 0; r < 16; ++r) {
                    int t = tid + r * 128;
                    int k = t >> 6, n = t & 63;
                    int cn = (n0 + n) < N  ? (n0 + n) : N - 1;
                    int ck = (k0 + k) < Kd ? (k0 + k) : Kd - 1;
                    vals[r] = Bb[(size_t)(cn * ldbi + ck)];
                }
                SCHED_FENCE();
                #pragma unroll
                for (int r = 0; r < 16; ++r) {
                    int t = tid + r * 128;
                    int k = t >> 6, n = t & 63;
                    float msk = ((n0 + n) < N && (k0 + k) < Kd) ? 1.f : 0.f;
                    Bs[k][n] = (bf16)(vals[r] * msk);
                }
            }
        } else {
            if (fullN && fullK && bvec) {
                #pragma unroll
                for (int r = 0; r < 4; ++r) {
                    int t  = tid + r * 128;
                    int k  = t >> 4;                 // 16 float4 per row of 64 n
                    int n4 = (t & 15) << 2;
                    const float4 v = *(const float4*)(Bb + (size_t)((k0 + k) * ldbi + n0 + n4));
                    v4bf p = { (bf16)v.x, (bf16)v.y, (bf16)v.z, (bf16)v.w };
                    *(v4bf*)&Bs[k][n4] = p;
                }
            } else {
                float vals[16];
                #pragma unroll
                for (int r = 0; r < 16; ++r) {
                    int t = tid + r * 128;
                    int k = t >> 6, n = t & 63;
                    int cn = (n0 + n) < N  ? (n0 + n) : N - 1;
                    int ck = (k0 + k) < Kd ? (k0 + k) : Kd - 1;
                    vals[r] = Bb[(size_t)(ck * ldbi + cn)];
                }
                SCHED_FENCE();
                #pragma unroll
                for (int r = 0; r < 16; ++r) {
                    int t = tid + r * 128;
                    int k = t >> 6, n = t & 63;
                    float msk = ((n0 + n) < N && (k0 + k) < Kd) ? 1.f : 0.f;
                    Bs[k][n] = (bf16)(vals[r] * msk);
                }
            }
        }
        __syncthreads();

        // ---- A fragment: lane = m (mod 16), documented K-half interleave ----
        v16bf a;
        {
            const int m   = wave * 16 + (lane & 15);
            const int klo = (lane < 16) ? 0 : 8;
            #pragma unroll
            for (int i = 0; i < 16; ++i) {
                int k = ((i < 8) ? i : i + 8) + klo;
                a[i] = As[m][k];
            }
        }
        // ---- B fragments: lane = n (mod 16), same K-half interleave as A ----
        #pragma unroll
        for (int j = 0; j < 4; ++j) {
            v16bf b;
            #pragma unroll
            for (int i = 0; i < 16; ++i) b[i] = Bs[((i < 8) ? i : i + 8) + ((lane < 16) ? 0 : 8)][j * 16 + (lane & 15)];
            acc[j] = __builtin_amdgcn_wmma_f32_16x16x32_bf16(
                false, a, false, b, (short)0, acc[j], false, false);
        }
        __syncthreads();
    }

    // ---- store: VGPR i holds row i (+8 for lanes 16..31), col = lane&15 ----
    #pragma unroll
    for (int j = 0; j < 4; ++j) {
        int n = n0 + j * 16 + (lane & 15);
        if (n >= N) continue;
        float bv = bias ? bias[n] : 0.f;
        int mbase = m0 + wave * 16 + ((lane < 16) ? 0 : 8);
        #pragma unroll
        for (int i = 0; i < 8; ++i) {
            int m = mbase + i;
            if (m < M) {
                float v = acc[j][i] * alpha + bv;
                if (relu) v = v > 0.f ? v : 0.f;
                Cb[(long long)m * ldc + n] = v;
            }
        }
    }
}

// ---------------------------------------------------------------------------
// Row softmax with optional causal + key-padding mask (decoder self-attn).
// ---------------------------------------------------------------------------
__global__ __launch_bounds__(256)
void softmax_rows(float* __restrict__ scores, int Slen, long long zStride,
                  int maskMode, const int* __restrict__ labels)
{
    const int t = blockIdx.x;
    const int z = blockIdx.y;
    const int b = z >> 3;  // z / H
    float* row = scores + (long long)z * zStride + (long long)t * Slen;
    __shared__ float red[256];
    const int tid = threadIdx.x;

    float mx = -3.4e38f;
    for (int s = tid; s < Slen; s += 256) {
        float v = row[s];
        if (maskMode && (s > t || labels[b * CL + s + 1] == 0)) v = -1e9f;
        mx = fmaxf(mx, v);
    }
    red[tid] = mx; __syncthreads();
    for (int off = 128; off; off >>= 1) { if (tid < off) red[tid] = fmaxf(red[tid], red[tid + off]); __syncthreads(); }
    mx = red[0]; __syncthreads();

    float sum = 0.f;
    for (int s = tid; s < Slen; s += 256) {
        float v = row[s];
        if (maskMode && (s > t || labels[b * CL + s + 1] == 0)) v = -1e9f;
        float e = expf(v - mx);
        row[s] = e;
        sum += e;
    }
    red[tid] = sum; __syncthreads();
    for (int off = 128; off; off >>= 1) { if (tid < off) red[tid] += red[tid + off]; __syncthreads(); }
    float inv = 1.f / red[0];
    for (int s = tid; s < Slen; s += 256) row[s] *= inv;
}

// ---------------------------------------------------------------------------
// out = LayerNorm(x + r) * g + b  (D = 768 = 3 * 256 exactly)
// ---------------------------------------------------------------------------
__global__ __launch_bounds__(256)
void ln_res(float* __restrict__ out, const float* __restrict__ x, const float* __restrict__ r,
            const float* __restrict__ g, const float* __restrict__ bb)
{
    const int row = blockIdx.x;
    const int tid = threadIdx.x;
    __shared__ float red[256];
    const float* xr = x + (long long)row * CD;
    const float* rr = r + (long long)row * CD;
    float* o = out + (long long)row * CD;

    float loc[3]; float s = 0.f;
    #pragma unroll
    for (int i = 0; i < 3; ++i) { float v = xr[tid + i * 256] + rr[tid + i * 256]; loc[i] = v; s += v; }
    red[tid] = s; __syncthreads();
    for (int off = 128; off; off >>= 1) { if (tid < off) red[tid] += red[tid + off]; __syncthreads(); }
    float mean = red[0] * (1.f / CD); __syncthreads();

    float vs = 0.f;
    #pragma unroll
    for (int i = 0; i < 3; ++i) { float d = loc[i] - mean; vs += d * d; }
    red[tid] = vs; __syncthreads();
    for (int off = 128; off; off >>= 1) { if (tid < off) red[tid] += red[tid + off]; __syncthreads(); }
    float inv = rsqrtf(red[0] * (1.f / CD) + 1e-5f);
    #pragma unroll
    for (int i = 0; i < 3; ++i) { int d = tid + i * 256; o[d] = (loc[i] - mean) * inv * g[d] + bb[d]; }
}

// ---------------------------------------------------------------------------
// VQ: per-token f32 argmin over 512 codes; gather emb row; accumulate loss.
// ---------------------------------------------------------------------------
__global__ __launch_bounds__(256)
void vq_kernel(const float* __restrict__ x, const float* __restrict__ cb,
               const float* __restrict__ emb, float* __restrict__ h, float* lossA)
{
    const int tok = blockIdx.x;
    const int tid = threadIdx.x;
    __shared__ float xs[CD];
    __shared__ float redv[256];
    __shared__ int   redi[256];

    const float* xr = x + (long long)tok * CD;
    for (int d = tid; d < CD; d += 256) xs[d] = xr[d];
    __syncthreads();

    float best = 3.4e38f; int bi = 0;
    for (int c = tid; c < CK; c += 256) {
        const float* cr = cb + (long long)c * CD;
        float dot = 0.f, cn = 0.f;
        for (int d = 0; d < CD; ++d) { float cv = cr[d]; dot += cv * xs[d]; cn += cv * cv; }
        float dist = cn - 2.f * dot;      // |x|^2 constant per row, irrelevant for argmin
        if (dist < best) { best = dist; bi = c; }
    }
    redv[tid] = best; redi[tid] = bi; __syncthreads();
    for (int off = 128; off; off >>= 1) {
        if (tid < off) {
            float ov = redv[tid + off]; int oi = redi[tid + off];
            if (ov < redv[tid] || (ov == redv[tid] && oi < redi[tid])) { redv[tid] = ov; redi[tid] = oi; }
        }
        __syncthreads();
    }
    const int idx = redi[0]; __syncthreads();

    const float* cr = cb + (long long)idx * CD;
    const float* er = emb + (long long)idx * CD;
    float ls = 0.f;
    for (int d = tid; d < CD; d += 256) {
        float dv = cr[d] - xs[d];
        ls += dv * dv;
        h[(long long)tok * CD + d] = er[d];
    }
    redv[tid] = ls; __syncthreads();
    for (int off = 128; off; off >>= 1) { if (tid < off) redv[tid] += redv[tid + off]; __syncthreads(); }
    if (tid == 0) atomicAdd(lossA, redv[0]);
}

__global__ __launch_bounds__(256)
void gather_decin(const int* __restrict__ labels, const float* __restrict__ tok, float* __restrict__ y)
{
    const int i = blockIdx.x;                // 0..783
    const int b = i / CT, t = i % CT;
    const float* src = tok + (long long)labels[b * CL + t] * CD;
    float* dst = y + (long long)i * CD;
    for (int d = threadIdx.x; d < CD; d += 256) dst[d] = src[d];
}

__global__ void zero_k(float* p) { if (threadIdx.x == 0) p[0] = 0.f; }

__global__ void finalize_k(const float* lossA, const int* __restrict__ labels, float* __restrict__ out)
{
    const int i = blockIdx.x * blockDim.x + threadIdx.x;
    const int logitsN = CB * CT * CV;        // 3,920,000
    if (i == 0) out[logitsN] = 0.25f * lossA[0] * (1.f / (float)(CB * CS * CD));
    if (i < CB * CT) {
        int b = i / CT, t = i % CT;
        out[logitsN + 1 + i] = (float)labels[b * CL + t + 1];
    }
}

// ---------------------------------------------------------------------------
extern "C" void kernel_launch(void* const* d_in, const int* in_sizes, int n_in,
                              void* d_out, int out_size, void* d_ws, size_t ws_size,
                              hipStream_t stream)
{
    (void)in_sizes; (void)n_in; (void)out_size; (void)ws_size;
    // setup_inputs() dict order:
    const float* features  = (const float*)d_in[0];
    const int*   labels    = (const int*)  d_in[1];
    const float* codebook  = (const float*)d_in[2];
    const float* emb_layer = (const float*)d_in[3];
    const float* tok_emb   = (const float*)d_in[4];
    const float* fc_W      = (const float*)d_in[5];
    const float* fc_b      = (const float*)d_in[6];
    const float* enc_Wqkv  = (const float*)d_in[7];
    const float* dec_sWqkv = (const float*)d_in[8];
    const float* dec_cWqkv = (const float*)d_in[9];
    const float* enc_bqkv  = (const float*)d_in[10];
    const float* enc_Wo    = (const float*)d_in[11];
    const float* enc_bo    = (const float*)d_in[12];
    const float* enc_ln1g  = (const float*)d_in[13];
    const float* enc_ln1b  = (const float*)d_in[14];
    const float* enc_W1    = (const float*)d_in[15];
    const float* enc_b1    = (const float*)d_in[16];
    const float* enc_W2    = (const float*)d_in[17];
    const float* enc_b2    = (const float*)d_in[18];
    const float* enc_ln2g  = (const float*)d_in[19];
    const float* enc_ln2b  = (const float*)d_in[20];
    const float* dec_sbqkv = (const float*)d_in[21];
    const float* dec_sWo   = (const float*)d_in[22];
    const float* dec_sbo   = (const float*)d_in[23];
    const float* dec_cbqkv = (const float*)d_in[24];
    const float* dec_cWo   = (const float*)d_in[25];
    const float* dec_cbo   = (const float*)d_in[26];
    const float* dec_ln1g  = (const float*)d_in[27];
    const float* dec_ln1b  = (const float*)d_in[28];
    const float* dec_ln2g  = (const float*)d_in[29];
    const float* dec_ln2b  = (const float*)d_in[30];
    const float* dec_ln3g  = (const float*)d_in[31];
    const float* dec_ln3b  = (const float*)d_in[32];
    const float* dec_W1    = (const float*)d_in[33];
    const float* dec_b1    = (const float*)d_in[34];
    const float* dec_W2    = (const float*)d_in[35];
    const float* dec_b2    = (const float*)d_in[36];

    float* out = (float*)d_out;

    // ---- workspace layout (floats) ----
    float* w    = (float*)d_ws;
    float* h    = w;                       // 4800*768          (encoder state / memory)
    float* qkv  = h    + 3686400;          // 4800*2304         (also cross K/V)
    float* big  = qkv  + 11059200;         // max(scores 11.52M, enc ffn 7.37M)
    float* tmp  = big  + 11520000;         // 4800*768
    float* tmp2 = tmp  + 3686400;          // 4800*768
    float* yb   = tmp2 + 3686400;          // 784*768           (decoder state)
    float* dtmp = yb   + 602112;           // 784*1536          (dec attn out / ffn mid)
    float* dtmp2= dtmp + 1204224;          // 784*768
    float* dqkv = dtmp2+ 602112;           // 784*2304          (also cross Q, ld 768)
    float* lossA= dqkv + 1806336;          // 1

    const int NTOK = CB * CS;              // 4800
    const int DTOK = CB * CT;              // 784
    const float sc = 1.0f / sqrtf((float)CHD);
    const long long ll0 = 0;

    auto GEMM = [&](const float* A, long long lda, long long sAo, long long sAi,
                    const float* Bp, long long ldb, long long sBo, long long sBi,
                    float* Cp, long long ldc, long long sCo, long long sCi,
                    int M, int N, int Kd, int batch, int inner,
                    const float* bias, float alpha, int relu, int transB) {
        dim3 g((unsigned)((N + BN - 1) / BN), (unsigned)((M + BM - 1) / BM), (unsigned)batch);
        hipLaunchKernelGGL(gemm_wmma, g, dim3(128), 0, stream,
                           A, lda, sAo, sAi, Bp, ldb, sBo, sBi, Cp, ldc, sCo, sCi,
                           M, N, Kd, inner, bias, alpha, relu, transB);
    };

    // ================= VQ =================
    hipLaunchKernelGGL(zero_k, dim3(1), dim3(32), 0, stream, lossA);
    hipLaunchKernelGGL(vq_kernel, dim3(NTOK), dim3(256), 0, stream,
                       features, codebook, emb_layer, h, lossA);

    // ================= Encoder =================
    for (int i = 0; i < 4; ++i) {
        const float* Wqkv = enc_Wqkv + (long long)i * 3 * CD * CD;
        const float* bqkv = enc_bqkv + (long long)i * 3 * CD;
        const float* Wo   = enc_Wo   + (long long)i * CD * CD;
        const float* bo   = enc_bo   + (long long)i * CD;
        const float* W1   = enc_W1   + (long long)i * CF * CD;
        const float* b1   = enc_b1   + (long long)i * CF;
        const float* W2   = enc_W2   + (long long)i * CD * CF;
        const float* b2   = enc_b2   + (long long)i * CD;

        GEMM(h, CD, ll0, ll0, Wqkv, CD, ll0, ll0, qkv, 3*CD, ll0, ll0,
             NTOK, 3*CD, CD, 1, 1, bqkv, 1.f, 0, 1);
        GEMM(qkv, 3*CD, (long long)CS*3*CD, CHD,
             qkv + CD, 3*CD, (long long)CS*3*CD, CHD,
             big, CS, (long long)CH*CS*CS, (long long)CS*CS,
             CS, CS, CHD, CB*CH, CH, nullptr, sc, 0, 1);
        hipLaunchKernelGGL(softmax_rows, dim3(CS, CB*CH), dim3(256), 0, stream,
                           big, CS, (long long)CS*CS, 0, labels);
        GEMM(big, CS, (long long)CH*CS*CS, (long long)CS*CS,
             qkv + 2*CD, 3*CD, (long long)CS*3*CD, CHD,
             tmp, CD, (long long)CS*CD, CHD,
             CS, CHD, CS, CB*CH, CH, nullptr, 1.f, 0, 0);
        GEMM(tmp, CD, ll0, ll0, Wo, CD, ll0, ll0, tmp2, CD, ll0, ll0,
             NTOK, CD, CD, 1, 1, bo, 1.f, 0, 1);
        hipLaunchKernelGGL(ln_res, dim3(NTOK), dim3(256), 0, stream,
                           h, h, tmp2, enc_ln1g + i*CD, enc_ln1b + i*CD);
        GEMM(h, CD, ll0, ll0, W1, CD, ll0, ll0, big, CF, ll0, ll0,
             NTOK, CF, CD, 1, 1, b1, 1.f, 1, 1);
        GEMM(big, CF, ll0, ll0, W2, CF, ll0, ll0, tmp, CD, ll0, ll0,
             NTOK, CD, CF, 1, 1, b2, 1.f, 0, 1);
        hipLaunchKernelGGL(ln_res, dim3(NTOK), dim3(256), 0, stream,
                           h, h, tmp, enc_ln2g + i*CD, enc_ln2b + i*CD);
    }
    // h is now the encoder memory (16 x 300 x 768)

    // ================= Decoder =================
    hipLaunchKernelGGL(gather_decin, dim3(DTOK), dim3(256), 0, stream, labels, tok_emb, yb);

    for (int i = 0; i < 4; ++i) {
        const float* sWqkv = dec_sWqkv + (long long)i * 3 * CD * CD;
        const float* sbqkv = dec_sbqkv + (long long)i * 3 * CD;
        const float* sWo   = dec_sWo   + (long long)i * CD * CD;
        const float* sbo   = dec_sbo   + (long long)i * CD;
        const float* cWqkv = dec_cWqkv + (long long)i * 3 * CD * CD;
        const float* cbqkv = dec_cbqkv + (long long)i * 3 * CD;
        const float* cWo   = dec_cWo   + (long long)i * CD * CD;
        const float* cbo   = dec_cbo   + (long long)i * CD;
        const float* W1    = dec_W1    + (long long)i * CF * CD;
        const float* b1    = dec_b1    + (long long)i * CF;
        const float* W2    = dec_W2    + (long long)i * CD * CF;
        const float* b2    = dec_b2    + (long long)i * CD;

        // --- masked self-attention ---
        GEMM(yb, CD, ll0, ll0, sWqkv, CD, ll0, ll0, dqkv, 3*CD, ll0, ll0,
             DTOK, 3*CD, CD, 1, 1, sbqkv, 1.f, 0, 1);
        GEMM(dqkv, 3*CD, (long long)CT*3*CD, CHD,
             dqkv + CD, 3*CD, (long long)CT*3*CD, CHD,
             big, CT, (long long)CH*CT*CT, (long long)CT*CT,
             CT, CT, CHD, CB*CH, CH, nullptr, sc, 0, 1);
        hipLaunchKernelGGL(softmax_rows, dim3(CT, CB*CH), dim3(256), 0, stream,
                           big, CT, (long long)CT*CT, 1, labels);
        GEMM(big, CT, (long long)CH*CT*CT, (long long)CT*CT,
             dqkv + 2*CD, 3*CD, (long long)CT*3*CD, CHD,
             dtmp, CD, (long long)CT*CD, CHD,
             CT, CHD, CT, CB*CH, CH, nullptr, 1.f, 0, 0);
        GEMM(dtmp, CD, ll0, ll0, sWo, CD, ll0, ll0, dtmp2, CD, ll0, ll0,
             DTOK, CD, CD, 1, 1, sbo, 1.f, 0, 1);
        hipLaunchKernelGGL(ln_res, dim3(DTOK), dim3(256), 0, stream,
                           yb, yb, dtmp2, dec_ln1g + i*CD, dec_ln1b + i*CD);

        // --- cross-attention (K/V from encoder memory h) ---
        float* qb = dqkv;                  // 784*768
        float* kb = qkv;                   // 4800*768
        float* vb = qkv + 3686400;         // 4800*768
        GEMM(yb, CD, ll0, ll0, cWqkv,            CD, ll0, ll0, qb, CD, ll0, ll0,
             DTOK, CD, CD, 1, 1, cbqkv,        1.f, 0, 1);
        GEMM(h,  CD, ll0, ll0, cWqkv + CD*CD,   CD, ll0, ll0, kb, CD, ll0, ll0,
             NTOK, CD, CD, 1, 1, cbqkv + CD,   1.f, 0, 1);
        GEMM(h,  CD, ll0, ll0, cWqkv + 2*CD*CD, CD, ll0, ll0, vb, CD, ll0, ll0,
             NTOK, CD, CD, 1, 1, cbqkv + 2*CD, 1.f, 0, 1);
        GEMM(qb, CD, (long long)CT*CD, CHD,
             kb, CD, (long long)CS*CD, CHD,
             big, CS, (long long)CH*CT*CS, (long long)CT*CS,
             CT, CS, CHD, CB*CH, CH, nullptr, sc, 0, 1);
        hipLaunchKernelGGL(softmax_rows, dim3(CT, CB*CH), dim3(256), 0, stream,
                           big, CS, (long long)CT*CS, 0, labels);
        GEMM(big, CS, (long long)CH*CT*CS, (long long)CT*CS,
             vb, CD, (long long)CS*CD, CHD,
             dtmp, CD, (long long)CT*CD, CHD,
             CT, CHD, CS, CB*CH, CH, nullptr, 1.f, 0, 0);
        GEMM(dtmp, CD, ll0, ll0, cWo, CD, ll0, ll0, dtmp2, CD, ll0, ll0,
             DTOK, CD, CD, 1, 1, cbo, 1.f, 0, 1);
        hipLaunchKernelGGL(ln_res, dim3(DTOK), dim3(256), 0, stream,
                           yb, yb, dtmp2, dec_ln2g + i*CD, dec_ln2b + i*CD);

        // --- FFN ---
        GEMM(yb, CD, ll0, ll0, W1, CD, ll0, ll0, dtmp, CF, ll0, ll0,
             DTOK, CF, CD, 1, 1, b1, 1.f, 1, 1);
        GEMM(dtmp, CF, ll0, ll0, W2, CF, ll0, ll0, dtmp2, CD, ll0, ll0,
             DTOK, CD, CF, 1, 1, b2, 1.f, 0, 1);
        hipLaunchKernelGGL(ln_res, dim3(DTOK), dim3(256), 0, stream,
                           yb, yb, dtmp2, dec_ln3g + i*CD, dec_ln3b + i*CD);
    }

    // ================= Vocab head + outputs =================
    GEMM(yb, CD, ll0, ll0, fc_W, CD, ll0, ll0, out, CV, ll0, ll0,
         DTOK, CV, CD, 1, 1, fc_b, 1.f, 0, 1);
    hipLaunchKernelGGL(finalize_k, dim3(4), dim3(256), 0, stream, lossA, labels, out);
}
